// RGTSR_49143015801113
// MI455X (gfx1250) — matrix operand order repaired and checked by reading, not compiled, not run on hardware
//
#include <hip/hip_runtime.h>
#include <hip/hip_bf16.h>

#define NN 100000
#define DD 128
#define BB 1024
#define DEGQ 256
#define EE (BB*DEGQ)
#define TOPK 20

// Async global->LDS path (gfx1250 GLOBAL_LOAD_ASYNC_TO_LDS_*, ASYNCcnt):
// guarded so the file still compiles on toolchains without the builtins.
#if defined(__has_builtin)
#if __has_builtin(__builtin_amdgcn_global_load_async_to_lds_b128) && __has_builtin(__builtin_amdgcn_s_wait_asynccnt)
#define USE_ASYNC 1
#endif
#endif
#ifndef USE_ASYNC
#define USE_ASYNC 0
#endif

typedef int v4i __attribute__((ext_vector_type(4)));
#if USE_ASYNC
// builtin signature (from clang diagnostic): arg0 = v4i in addrspace(1) [global],
// arg1 = LDS destination, arg2/arg3 = imm offset / cpol.
typedef __attribute__((address_space(1))) v4i GV4;
typedef __attribute__((address_space(3))) v4i LV4;
#endif

typedef __attribute__((ext_vector_type(16))) __bf16 v16bf;
typedef __attribute__((ext_vector_type(8)))  float  v8f;

union V16BF { v16bf v; __bf16 e[16]; };
union V8F   { v8f  v; float e[8]; };

__device__ __forceinline__ __bf16 f2bf(float x) { return (__bf16)x; }

// A-operand element i (of 16 per lane) -> K offset within a 32-deep tile,
// in this file's contiguous order k = 16*half + i (B packed the same): lanes 0-15 rows, half selects K group.
__device__ __forceinline__ int a_koff(int i, int half) {
  return half*16 + i;
}

// ---------------- M = Wq^T * Wk  (512x512, bf16 WMMA, f32 accum) -------------
__global__ void k_gemm_M(const float* __restrict__ Wq, const float* __restrict__ Wk,
                         float* __restrict__ M) {
  int lane = threadIdx.x & 31, w = threadIdx.x >> 5;
  int r = lane & 15, half = lane >> 4;
  int m0 = blockIdx.x * 16;
  int n0 = blockIdx.y * 128 + w * 16;
  V8F c;
#pragma unroll
  for (int i = 0; i < 8; i++) c.e[i] = 0.f;
  for (int k0 = 0; k0 < 512; k0 += 32) {
    V16BF a, b;
#pragma unroll
    for (int i = 0; i < 16; i++) {
      int ka = k0 + a_koff(i, half);
      a.e[i] = f2bf(Wq[ka*512 + m0 + r]);      // A = Wq^T : A[m][k] = Wq[k][m]
      int kb = k0 + half*16 + i;               // B 32x16 layout
      b.e[i] = f2bf(Wk[kb*512 + n0 + r]);
    }
    c.v = __builtin_amdgcn_wmma_f32_16x16x32_bf16(false, a.v, false, b.v,
                                                  (short)0, c.v, false, false);
  }
#pragma unroll
  for (int v = 0; v < 8; v++)
    M[(m0 + v + 8*half)*512 + n0 + r] = c.e[v];
}

// -------- pack B-operands into WMMA lane-order bf16 (one 32B load / tile) ----
// layout: idx = ((tileN*4 + tileK)*32 + lane)*16 + i
__global__ void k_pack(const float* __restrict__ M, const float* __restrict__ Wlin,
                       __bf16* __restrict__ W22s, __bf16* __restrict__ Wnodes,
                       __bf16* __restrict__ WlinTs) {
  int idx = blockIdx.x * blockDim.x + threadIdx.x;
  int i = idx & 15, lane = (idx >> 4) & 31;
  int tk = (idx >> 9) & 3;
  int half = lane >> 4, r = lane & 15;
  int k = tk*32 + half*16 + i;                 // B element K
  if (idx < 49152) {                           // Wnode: 128 x 384 -> [M11|M12|M21^T]
    int tn = idx >> 11;
    int cc = tn*16 + r;
    float v = (cc < 256) ? M[k*512 + cc] : M[(128 + cc - 256)*512 + k];
    Wnodes[idx] = f2bf(v);
  } else if (idx < 49152 + 16384) {            // W22 = M[128:256,128:256]
    int j = idx - 49152;
    int tn = j >> 11;
    int n = tn*16 + r;
    W22s[j] = f2bf(M[(128 + k)*512 + 128 + n]);
  } else if (idx < 49152 + 32768) {            // WlinT[k][o] = W_lin[o][k]
    int j = idx - 49152 - 16384;
    int tn = j >> 11;
    int n = tn*16 + r;
    WlinTs[j] = f2bf(Wlin[n*128 + k]);
  }
}

// -------- per-query vectors w (dot hvi), x (dot hvj), y (dot rel), scalar s --
__global__ void k_query(const float* __restrict__ M, const float* __restrict__ qsrc,
                        const float* __restrict__ qrel, float* __restrict__ qvec,
                        float* __restrict__ qs) {
  __shared__ float u[256];
  __shared__ float red[128];
  int g = blockIdx.x, t = threadIdx.x;   // 128 threads
  u[t]       = qsrc[g*128 + t];
  u[128 + t] = qrel[g*128 + t];
  __syncthreads();
  float wv = 0.f, xv = 0.f, yv = 0.f;
  for (int b = 0; b < 256; b++) {
    float ub = u[b];
    wv += M[t*512 + 256 + b] * ub;               // M13|M14 row t
    yv += M[(128 + t)*512 + 256 + b] * ub;       // M23|M24 row t
    xv += M[(256 + b)*512 + t] * ub;             // M31|M41 col t
    yv += M[(256 + b)*512 + 128 + t] * ub;       // M32|M42 col t
  }
  qvec[g*384 + t]       = wv;
  qvec[g*384 + 128 + t] = xv;
  qvec[g*384 + 256 + t] = yv;
  float sp = 0.f;
  for (int rr = t; rr < 256; rr += 128) {
    float acc = 0.f;
    for (int b = 0; b < 256; b++) acc += M[(256 + rr)*512 + 256 + b] * u[b];
    sp += u[rr] * acc;
  }
  red[t] = sp; __syncthreads();
  for (int s = 64; s > 0; s >>= 1) { if (t < s) red[t] += red[t + s]; __syncthreads(); }
  if (t == 0) qs[g] = red[0];
}

// ------------- T = node_repr @ [M11|M12|M21^T]  (N x 384, bf16 out) ----------
__global__ void k_nodeT(const float* __restrict__ nrep, const __bf16* __restrict__ Wnodes,
                        __bf16* __restrict__ T) {
  __shared__ v16bf s_a[128];                   // 4 K-tiles x 32 lanes, swizzled
  int tid = threadIdx.x;
  int m0 = blockIdx.x * 16;
  __bf16* ap = (__bf16*)s_a;
  for (int idx = tid; idx < 2048; idx += 256) {
    int i = idx & 15, lane = (idx >> 4) & 31, tk = idx >> 9;
    int half = lane >> 4, r = lane & 15;
    int k = tk*32 + a_koff(i, half);
    ap[idx] = f2bf(nrep[(size_t)(m0 + r)*128 + k]);
  }
  __syncthreads();
  int lane = tid & 31, w = tid >> 5, r = lane & 15, half = lane >> 4;
  for (int tt = 0; tt < 3; tt++) {
    int tn = w + tt*8;
    V8F c;
#pragma unroll
    for (int i = 0; i < 8; i++) c.e[i] = 0.f;
#pragma unroll
    for (int tk = 0; tk < 4; tk++) {
      v16bf a = s_a[tk*32 + lane];
      v16bf b = *(const v16bf*)(Wnodes + ((size_t)(tn*4 + tk)*32 + lane)*16);
      c.v = __builtin_amdgcn_wmma_f32_16x16x32_bf16(false, a, false, b,
                                                    (short)0, c.v, false, false);
    }
#pragma unroll
    for (int v = 0; v < 8; v++)
      T[(size_t)(m0 + v + 8*half)*384 + tn*16 + r] = f2bf(c.e[v]);
  }
}

// ----------------------------- zero init -------------------------------------
__global__ void k_init(float* __restrict__ agg, float* __restrict__ outScore,
                       int* __restrict__ appears, float* __restrict__ segsum,
                       unsigned* __restrict__ segkey) {
  int idx = blockIdx.x * 256 + threadIdx.x;
  if (idx < NN*128) agg[idx] = 0.f;
  if (idx < NN) { outScore[idx] = 0.f; appears[idx] = 0; segsum[idx] = 0.f; segkey[idx] = 0u; }
}

// -------- fused edge kernel: rel@M22 (WMMA) + all bilinear dot terms ---------
__global__ void k_edge(const float* __restrict__ rel, const float* __restrict__ nrep,
                       const int* __restrict__ edges, const __bf16* __restrict__ W22s,
                       const __bf16* __restrict__ T, const float* __restrict__ qvec,
                       const float* __restrict__ qs, float* __restrict__ logits) {
  __shared__ __align__(16) float relf[16*128];
  __shared__ v16bf s_a[128];
  __shared__ float qw[384];
  __shared__ float accrow[16];
  __shared__ int viA[16], vjA[16];
  int tid = threadIdx.x;
  int e0 = blockIdx.x * 16;
  int g = e0 >> 8;                              // 16 edges per block share one query
#if USE_ASYNC
  // 8KB rel tile: async DMA global -> LDS, no VGPR round trip (ASYNCcnt path)
  for (int cI = tid; cI < 512; cI += 256) {
    const float* gp = rel + (size_t)e0*128 + cI*4;
    float* lp = relf + cI*4;
    __builtin_amdgcn_global_load_async_to_lds_b128(
        (GV4*)(unsigned long long)(size_t)gp,
        (LV4*)(unsigned)(size_t)lp, 0, 0);
  }
#else
  for (int idx = tid; idx < 2048; idx += 256)
    relf[idx] = rel[(size_t)e0*128 + idx];      // 16 contiguous rel rows
#endif
  if (tid < 384) qw[tid] = qvec[g*384 + tid];
  if (tid < 16) {
    accrow[tid] = 0.f;
    viA[tid] = edges[(size_t)(e0 + tid)*8 + 6];
    vjA[tid] = edges[(size_t)(e0 + tid)*8 + 7];
  }
#if USE_ASYNC
  __builtin_amdgcn_s_wait_asynccnt(0);
#endif
  __syncthreads();
  // Prefetch the random-index gather lines (node_repr / T rows) so they are
  // L0/L2 resident by the time the post-WMMA epilogue runs.
  {
    int m = tid >> 4, part = tid & 15;          // 16 edges x 16 cache lines
    const char* p;
    if (part < 4)       p = (const char*)(nrep + (size_t)viA[m]*128) + part*128;
    else if (part < 8)  p = (const char*)(nrep + (size_t)vjA[m]*128) + (part - 4)*128;
    else if (part < 12) p = (const char*)(T + (size_t)viA[m]*384) + (part - 8)*128;
    else                p = (const char*)(T + (size_t)vjA[m]*384 + 256) + (part - 12)*64;
    __builtin_prefetch(p, 0, 1);
  }
  __bf16* ap = (__bf16*)s_a;
  for (int idx = tid; idx < 2048; idx += 256) {
    int i = idx & 15, lane = (idx >> 4) & 31, tk = idx >> 9;
    int half = lane >> 4, r = lane & 15;
    ap[idx] = f2bf(relf[r*128 + tk*32 + a_koff(i, half)]);
  }
  __syncthreads();
  int lane = tid & 31, w = tid >> 5, r = lane & 15, half = lane >> 4;
  V8F c;
#pragma unroll
  for (int i = 0; i < 8; i++) c.e[i] = 0.f;
#pragma unroll
  for (int tk = 0; tk < 4; tk++) {
    v16bf a = s_a[tk*32 + lane];
    v16bf b = *(const v16bf*)(W22s + ((size_t)(w*4 + tk)*32 + lane)*16);
    c.v = __builtin_amdgcn_wmma_f32_16x16x32_bf16(false, a, false, b,
                                                  (short)0, c.v, false, false);
  }
#pragma unroll
  for (int v = 0; v < 8; v++) {                 // rel^T M22 rel diagonal dot
    int m = v + 8*half;
    atomicAdd(&accrow[m], c.e[v] * relf[m*128 + w*16 + r]);
  }
  __syncthreads();
  float sg = qs[g];
  for (int t2 = 0; t2 < 2; t2++) {              // each wave finishes 2 edges
    int m = w + t2*8;
    int vi = viA[m], vj = vjA[m];
    const float* hvi = nrep + (size_t)vi*128;
    const float* hvj = nrep + (size_t)vj*128;
    const __bf16* Ti = T + (size_t)vi*384;      // [P | A]
    const __bf16* Tj = T + (size_t)vj*384 + 256;// [G]
    float p = 0.f;
#pragma unroll
    for (int q = 0; q < 4; q++) {
      int cc = lane*4 + q;
      float hi = hvi[cc], hj = hvj[cc], re = relf[m*128 + cc];
      p += hi*qw[cc] + hj*qw[128 + cc] + re*qw[256 + cc];
      p += (float)Ti[cc]*hj + ((float)Ti[128 + cc] + (float)Tj[cc])*re;
    }
#pragma unroll
    for (int off = 16; off > 0; off >>= 1) p += __shfl_xor(p, off, 32);
    if (lane == 0) logits[e0 + m] = p + accrow[m] + sg;
  }
}

// --------------- segment softmax over vi (monotone-uint max trick) -----------
__global__ void k_segmax(const float* __restrict__ logits, const int* __restrict__ edges,
                         unsigned* __restrict__ segkey) {
  int e = blockIdx.x*256 + threadIdx.x;
  int vi = edges[(size_t)e*8 + 6];
  unsigned u = __float_as_uint(logits[e]);
  unsigned key = (u & 0x80000000u) ? ~u : (u | 0x80000000u);
  atomicMax(&segkey[vi], key);
}

__global__ void k_segexp(float* __restrict__ logits, const int* __restrict__ edges,
                         const unsigned* __restrict__ segkey, float* __restrict__ segsum) {
  int e = blockIdx.x*256 + threadIdx.x;
  int vi = edges[(size_t)e*8 + 6];
  unsigned key = segkey[vi];
  unsigned u = (key & 0x80000000u) ? (key & 0x7fffffffu) : ~key;
  float m = __uint_as_float(u);
  float ex = __expf(logits[e] - m);
  logits[e] = ex;                                // in-place: logits -> exp
  atomicAdd(&segsum[vi], ex);
}

__global__ void k_smts(const float* __restrict__ ex, const int* __restrict__ edges,
                       const float* __restrict__ segsum, const float* __restrict__ nscore,
                       float* __restrict__ sm, float* __restrict__ ts) {
  int e = blockIdx.x*256 + threadIdx.x;
  int vi = edges[(size_t)e*8 + 6];
  float v = ex[e] / segsum[vi];
  sm[e] = v;
  ts[e] = v * nscore[vi];
}

// ------------------------ top-20 per query row (256 wide) --------------------
__global__ void k_topk(const float* __restrict__ ts, int* __restrict__ topk) {
  __shared__ float v[256];
  __shared__ float rv[256];
  __shared__ int ri[256];
  int g = blockIdx.x, t = threadIdx.x;
  v[t] = ts[g*256 + t];
  __syncthreads();
  for (int k = 0; k < TOPK; k++) {
    rv[t] = v[t]; ri[t] = t; __syncthreads();
    for (int s = 128; s > 0; s >>= 1) {
      if (t < s) {
        if (rv[t+s] > rv[t] || (rv[t+s] == rv[t] && ri[t+s] < ri[t])) {
          rv[t] = rv[t+s]; ri[t] = ri[t+s];
        }
      }
      __syncthreads();
    }
    if (t == 0) { topk[g*TOPK + k] = g*256 + ri[0]; v[ri[0]] = -3.4e38f; }
    __syncthreads();
  }
}

// -------------------- scatter: scores + repr aggregation ---------------------
__global__ void k_scatter(const int* __restrict__ topk, const int* __restrict__ edges,
                          const float* __restrict__ sm, const float* __restrict__ nscore,
                          const float* __restrict__ nrep, float* __restrict__ outScore,
                          float* __restrict__ agg, int* __restrict__ appears) {
  int p = blockIdx.x, t = threadIdx.x;          // 128 threads
  int e = topk[p];
  int vi = edges[(size_t)e*8 + 6], vj = edges[(size_t)e*8 + 7];
  float s = sm[e];
  if (t == 0) { atomicAdd(&outScore[vj], s * nscore[vi]); appears[vi] = 1; }
  atomicAdd(&agg[(size_t)vi*128 + t], s * nrep[(size_t)vj*128 + t]);
}

// ------------- final: leaky_relu((agg + coef*h) @ Wlin^T + b) ----------------
__global__ void k_final(const float* __restrict__ agg, const float* __restrict__ nrep,
                        const int* __restrict__ appears, const __bf16* __restrict__ WlinTs,
                        const float* __restrict__ blin, float* __restrict__ outRep) {
  __shared__ v16bf s_a[128];
  int tid = threadIdx.x;
  int m0 = blockIdx.x * 16;
  __bf16* ap = (__bf16*)s_a;
  for (int idx = tid; idx < 2048; idx += 256) {
    int i = idx & 15, lane = (idx >> 4) & 31, tk = idx >> 9;
    int half = lane >> 4, rr = lane & 15;
    int k = tk*32 + a_koff(i, half);
    int mg = m0 + rr;
    float coef = appears[mg] ? 0.f : 1.f;       // RATIO == 0
    ap[idx] = f2bf(agg[(size_t)mg*128 + k] + coef * nrep[(size_t)mg*128 + k]);
  }
  __syncthreads();
  int lane = tid & 31, w = tid >> 5, r = lane & 15, half = lane >> 4;
  V8F c;
#pragma unroll
  for (int i = 0; i < 8; i++) c.e[i] = 0.f;
#pragma unroll
  for (int tk = 0; tk < 4; tk++) {
    v16bf a = s_a[tk*32 + lane];
    v16bf b = *(const v16bf*)(WlinTs + ((size_t)(w*4 + tk)*32 + lane)*16);
    c.v = __builtin_amdgcn_wmma_f32_16x16x32_bf16(false, a, false, b,
                                                  (short)0, c.v, false, false);
  }
#pragma unroll
  for (int v = 0; v < 8; v++) {
    int mg = m0 + v + 8*half, n = w*16 + r;
    float x = c.e[v] + blin[n];
    outRep[(size_t)mg*128 + n] = (x >= 0.f) ? x : 0.01f*x;
  }
}

extern "C" void kernel_launch(void* const* d_in, const int* in_sizes, int n_in,
                              void* d_out, int out_size, void* d_ws, size_t ws_size,
                              hipStream_t stream) {
  (void)in_sizes; (void)n_in; (void)out_size; (void)ws_size;
  const float* nscore = (const float*)d_in[0];
  const float* nrep   = (const float*)d_in[1];
  const float* rel    = (const float*)d_in[2];
  const float* qsrc   = (const float*)d_in[3];
  const float* qrel   = (const float*)d_in[4];
  const float* Wq     = (const float*)d_in[5];
  const float* Wk     = (const float*)d_in[6];
  const float* Wlin   = (const float*)d_in[7];
  const float* blin   = (const float*)d_in[8];
  const int*   edges  = (const int*)d_in[9];
  float* outScore = (float*)d_out;
  float* outRep   = outScore + NN;

  char* ws = (char*)d_ws;
  size_t off = 0;
  auto alloc = [&](size_t bytes) -> char* {
    char* p = ws + off;
    off += (bytes + 255) & ~(size_t)255;
    return p;
  };
  float*    M      = (float*)   alloc((size_t)512*512*4);
  __bf16*   Wnodes = (__bf16*)  alloc((size_t)49152*2);
  __bf16*   W22s   = (__bf16*)  alloc((size_t)16384*2);
  __bf16*   WlinTs = (__bf16*)  alloc((size_t)16384*2);
  float*    qvec   = (float*)   alloc((size_t)BB*384*4);
  float*    qs     = (float*)   alloc((size_t)BB*4);
  __bf16*   T      = (__bf16*)  alloc((size_t)NN*384*2);
  float*    logits = (float*)   alloc((size_t)EE*4);
  float*    smv    = (float*)   alloc((size_t)EE*4);
  float*    tsv    = (float*)   alloc((size_t)EE*4);
  unsigned* segkey = (unsigned*)alloc((size_t)NN*4);
  float*    segsum = (float*)   alloc((size_t)NN*4);
  float*    agg    = (float*)   alloc((size_t)NN*128*4);
  int*      appears= (int*)     alloc((size_t)NN*4);
  int*      topk   = (int*)     alloc((size_t)BB*TOPK*4);

  k_init   <<<(NN*128 + 255)/256, 256, 0, stream>>>(agg, outScore, appears, segsum, segkey);
  k_gemm_M <<<dim3(32, 4), 256, 0, stream>>>(Wq, Wk, M);
  k_pack   <<<320, 256, 0, stream>>>(M, Wlin, W22s, Wnodes, WlinTs);
  k_query  <<<BB, 128, 0, stream>>>(M, qsrc, qrel, qvec, qs);
  k_nodeT  <<<NN/16, 256, 0, stream>>>(nrep, Wnodes, T);
  k_edge   <<<EE/16, 256, 0, stream>>>(rel, nrep, edges, W22s, T, qvec, qs, logits);
  k_segmax <<<EE/256, 256, 0, stream>>>(logits, edges, segkey);
  k_segexp <<<EE/256, 256, 0, stream>>>(logits, edges, segkey, segsum);
  k_smts   <<<EE/256, 256, 0, stream>>>(logits, edges, segsum, nscore, smv, tsv);
  k_topk   <<<BB, 256, 0, stream>>>(tsv, topk);
  k_scatter<<<BB*TOPK, 128, 0, stream>>>(topk, edges, smv, nscore, nrep, outScore, agg, appears);
  k_final  <<<NN/16, 256, 0, stream>>>(agg, nrep, appears, WlinTs, blin, outRep);
}
